// DeformableAttentionRGB_20486994002242
// MI455X (gfx1250) — matrix operand order repaired and checked by reading, not compiled, not run on hardware
//
#include <hip/hip_runtime.h>
#include <hip/hip_bf16.h>

// ---------------------------------------------------------------------------
// Types for CDNA5 WMMA (gfx1250, wave32)
// ---------------------------------------------------------------------------
typedef __bf16 bf16_t;
typedef __attribute__((ext_vector_type(16))) __bf16 v16bf;
typedef __attribute__((ext_vector_type(8)))  float  v8f;

struct __align__(16) U128 { unsigned int x, y, z, w; };
struct __align__(16) AB2  { U128 a, b; };

// A-fragment (16x32 bf16, M x K):  lane: m = lane&15, hi = lane>>4
//   element j -> k = k0 + j + hi*16
//   => one contiguous 32B region at (k0 + hi*16)
__device__ inline v16bf ld_fragA(const bf16_t* p /* = W + m*KTOT + k0 + hi*16 */) {
    const U128* q = (const U128*)p;
    AB2 t;
    t.a = q[0];
    t.b = q[1];   // +16 bytes = +8 bf16
    return __builtin_bit_cast(v16bf, t);
}

// B-fragment (32x16 bf16, K x N): lane: n = lane&15, hi = lane>>4
//   element j -> k = k0 + j + hi*16   => one contiguous 32B region
__device__ inline v16bf ld_fragB(const bf16_t* p /* = &tile[n][k0 + hi*16] */) {
    const U128* q = (const U128*)p;
    AB2 t;
    t.a = q[0];
    t.b = q[1];
    return __builtin_bit_cast(v16bf, t);
}

__device__ inline float sigmoidf_(float x) { return 1.f / (1.f + __expf(-x)); }

// ---------------------------------------------------------------------------
// Problem constants
// ---------------------------------------------------------------------------
#define BATCH 4
#define CC    64
#define HH    192
#define WW    192
#define HWX   (HH * WW)           // 36864
#define PXG   16                  // pixels per group (one row segment)
#define GPR   (WW / PXG)          // 12 groups per row
#define NGRP  (BATCH * HH * GPR)  // 9216 workgroups

// ---------------------------------------------------------------------------
// Weight packing: src OIHW f32 -> dst bf16 [co][t*CIN + c]
// ---------------------------------------------------------------------------
__global__ void pack_conv_kernel(const float* __restrict__ src, bf16_t* __restrict__ dst,
                                 int COUT, int CIN) {
    int n = COUT * CIN * 9;
    for (int i = blockIdx.x * blockDim.x + threadIdx.x; i < n; i += gridDim.x * blockDim.x) {
        int co  = i / (CIN * 9);
        int rem = i - co * (CIN * 9);
        int t   = rem / CIN;
        int c   = rem - t * CIN;
        dst[i] = (bf16_t)src[(co * CIN + c) * 9 + t];
    }
}

// offset(18) + mask(9) fused -> 32-row A matrix [co][t*128 + c] (rows 27..31 zero)
__global__ void pack_offmask_kernel(const float* __restrict__ offw,
                                    const float* __restrict__ maskw,
                                    bf16_t* __restrict__ dst) {
    int n = 32 * 1152;
    for (int i = blockIdx.x * blockDim.x + threadIdx.x; i < n; i += gridDim.x * blockDim.x) {
        int co  = i / 1152;
        int rem = i - co * 1152;
        int t   = rem / 128;
        int c   = rem - t * 128;
        float v = 0.f;
        if (co < 18)      v = offw[(co * 128 + c) * 9 + t];
        else if (co < 27) v = maskw[((co - 18) * 128 + c) * 9 + t];
        dst[i] = (bf16_t)v;
    }
}

// deform weights -> 9 per-tap A matrices: dst[t][co][c] = w[co][c][t]
__global__ void pack_deform_kernel(const float* __restrict__ src, bf16_t* __restrict__ dst) {
    int n = 9 * 64 * 64;
    for (int i = blockIdx.x * blockDim.x + threadIdx.x; i < n; i += gridDim.x * blockDim.x) {
        int t   = i / 4096;
        int rem = i - t * 4096;
        int co  = rem / 64;
        int c   = rem - co * 64;
        dst[i] = (bf16_t)src[(co * 64 + c) * 9 + t];
    }
}

// ---------------------------------------------------------------------------
// Implicit-GEMM 3x3 conv via WMMA bf16.
//   IN_MODE: 0 = single input (CIN ch), 1 = concat(in0[64], in1[64]),
//            2 = elementwise add in0+in1 (both CIN ch)
//   ACT:     0 = none, 1 = relu
// Block = 128 threads (4 waves). Each wave: 16 outch x 16 px tile.
// ---------------------------------------------------------------------------
template <int CIN, int COUT, int ACT, int IN_MODE>
__global__ __launch_bounds__(128)
void conv3x3_wmma_kernel(const float* __restrict__ in0, const float* __restrict__ in1,
                         const bf16_t* __restrict__ Wr,
                         const float* __restrict__ bias0, const float* __restrict__ bias1,
                         int bsplit, float* __restrict__ out) {
    constexpr int SC   = CIN + 8;        // padded channel stride (keeps 16B align)
    constexpr int KTOT = 9 * CIN;

    __shared__ __align__(16) bf16_t patch[3 * 18 * SC];

    const int g   = blockIdx.x;
    const int b   = g / (HH * GPR);
    const int rem = g - b * (HH * GPR);
    const int h   = rem / GPR;
    const int w0  = (rem - h * GPR) * PXG;

    // --- stage 3x18xCIN input patch into LDS as bf16 (channel-innermost) ---
    for (int idx = threadIdx.x; idx < 3 * 18 * CIN; idx += 128) {
        int c   = idx / 54;
        int pos = idx - c * 54;
        int row = pos / 18;
        int col = pos - row * 18;
        int hh  = h + row - 1;
        int ww  = w0 + col - 1;
        float v = 0.f;
        if (hh >= 0 && hh < HH && ww >= 0 && ww < WW) {
            if (IN_MODE == 1) {
                const float* src = (c < 64) ? in0 : in1;
                int cc = (c < 64) ? c : c - 64;
                v = src[((size_t)(b * 64 + cc) * HH + hh) * WW + ww];
            } else if (IN_MODE == 2) {
                size_t off = ((size_t)(b * CIN + c) * HH + hh) * WW + ww;
                v = in0[off] + in1[off];
            } else {
                v = in0[((size_t)(b * CIN + c) * HH + hh) * WW + ww];
            }
        }
        patch[pos * SC + c] = (bf16_t)v;
    }
    __syncthreads();

    const int lane  = threadIdx.x & 31;
    const int wave  = threadIdx.x >> 5;
    const int laneM = lane & 15;
    const int hi    = lane >> 4;
    const int co0   = wave * 16;

    if (co0 < COUT) {
        v8f acc = {0.f, 0.f, 0.f, 0.f, 0.f, 0.f, 0.f, 0.f};
        const bf16_t* Wbase = Wr + (size_t)(co0 + laneM) * KTOT + hi * 16;
#pragma unroll
        for (int t = 0; t < 9; ++t) {
            const int dy = t / 3, dx = t - dy * 3;
            const bf16_t* brow = &patch[(dy * 18 + laneM + dx) * SC + hi * 16];
            const bf16_t* arow = Wbase + t * CIN;
#pragma unroll
            for (int c0 = 0; c0 < CIN; c0 += 32) {
                v16bf af = ld_fragA(arow + c0);
                v16bf bf = ld_fragB(brow + c0);
                acc = __builtin_amdgcn_wmma_f32_16x16x32_bf16(
                        false, af, false, bf, (short)0, acc, false, false);
            }
        }
#pragma unroll
        for (int r = 0; r < 8; ++r) {
            int co = co0 + r + hi * 8;
            if (co < COUT) {
                float bv = (co < bsplit) ? bias0[co] : bias1[co - bsplit];
                float v  = acc[r] + bv;
                if (ACT == 1) v = fmaxf(v, 0.f);
                out[(((size_t)b * COUT + co) * HH + h) * WW + w0 + laneM] = v;
            }
        }
    }
}

// ---------------------------------------------------------------------------
// Channel attention: per-(b,c) mean of res2
// ---------------------------------------------------------------------------
__global__ __launch_bounds__(256)
void ca_reduce_kernel(const float* __restrict__ res2, float* __restrict__ means) {
    const int bc = blockIdx.x;                 // b*64 + c
    const float* p = res2 + (size_t)bc * HWX;
    float s = 0.f;
    for (int i = threadIdx.x; i < HWX; i += 256) s += p[i];
    __shared__ float red[256];
    red[threadIdx.x] = s;
    __syncthreads();
    for (int off = 128; off > 0; off >>= 1) {
        if (threadIdx.x < off) red[threadIdx.x] += red[threadIdx.x + off];
        __syncthreads();
    }
    if (threadIdx.x == 0) means[bc] = red[0] * (1.f / (float)HWX);
}

// tiny squeeze-excite MLP: y = sigmoid(W2 relu(W1 m + b1) + b2)
__global__ __launch_bounds__(256)
void ca_mlp_kernel(const float* __restrict__ means,
                   const float* __restrict__ w1, const float* __restrict__ b1,
                   const float* __restrict__ w2, const float* __restrict__ b2,
                   float* __restrict__ y) {
    const int t = threadIdx.x;        // b*64 + c
    const int b = t >> 6, c = t & 63;
    const float* m = means + b * 64;
    float hbuf[4];
#pragma unroll
    for (int r = 0; r < 4; ++r) {
        float s = b1[r];
        for (int k = 0; k < 64; ++k) s += w1[r * 64 + k] * m[k];
        hbuf[r] = fmaxf(s, 0.f);
    }
    float s = b2[c];
#pragma unroll
    for (int r = 0; r < 4; ++r) s += w2[c * 4 + r] * hbuf[r];
    y[t] = sigmoidf_(s);
}

// out1 = res2 * y[b,c] + x
__global__ __launch_bounds__(256)
void rcab_finish_kernel(const float* __restrict__ res2, const float* __restrict__ x,
                        const float* __restrict__ y, float* __restrict__ out1) {
    int i = blockIdx.x * 256 + threadIdx.x;
    if (i < BATCH * CC * HWX) {
        int bc = i / HWX;
        out1[i] = res2[i] * y[bc] + x[i];
    }
}

// ---------------------------------------------------------------------------
// Deformable conv: per tap, bilinear-gather a 16px x 64ch bf16 tile, then
// accumulate the 64x64 einsum with WMMA. Block = 128 (4 waves), 16 px/group.
// ---------------------------------------------------------------------------
__global__ __launch_bounds__(128)
void deform_wmma_kernel(const float* __restrict__ out1, const float* __restrict__ om,
                        const bf16_t* __restrict__ Wd, float* __restrict__ out2) {
    __shared__ float soy[9][16], sox[9][16], smk[9][16];
    __shared__ __align__(16) bf16_t val[16 * 72];   // [px][c], stride 72 bf16

    const int g   = blockIdx.x;
    const int b   = g / (HH * GPR);
    const int rem = g - b * (HH * GPR);
    const int h   = rem / GPR;
    const int w0  = (rem - h * GPR) * PXG;
    const int tid = threadIdx.x;

    // offsets (raw) + masks (2*sigmoid) for this pixel group
    for (int idx = tid; idx < 27 * 16; idx += 128) {
        int ch = idx >> 4, px = idx & 15;
        float v = om[(((size_t)b * 27 + ch) * HH + h) * WW + w0 + px];
        if (ch < 18) {
            int k = ch >> 1;
            if (ch & 1) sox[k][px] = v; else soy[k][px] = v;
        } else {
            smk[ch - 18][px] = 2.f * sigmoidf_(v);
        }
    }
    __syncthreads();

    const int lane  = tid & 31;
    const int wave  = tid >> 5;
    const int laneM = lane & 15;
    const int hi    = lane >> 4;
    const int px_b  = tid >> 3;          // pixel this thread samples for
    const int c_b   = (tid & 7) * 8;     // 8-channel slice

    v8f acc = {0.f, 0.f, 0.f, 0.f, 0.f, 0.f, 0.f, 0.f};

    for (int t = 0; t < 9; ++t) {
        // ---- build masked bilinear sample tile val[px][c] ----
        {
            const int ky = t / 3 - 1, kx = t % 3 - 1;
            float ys = (float)(h + ky) + soy[t][px_b];
            float xs = (float)(w0 + px_b + kx) + sox[t][px_b];
            float y0f = floorf(ys), x0f = floorf(xs);
            float wy1 = ys - y0f, wy0 = 1.f - wy1;
            float wx1 = xs - x0f, wx0 = 1.f - wx1;
            int y0 = (int)y0f, x0 = (int)x0f;
            int y1 = y0 + 1,   x1 = x0 + 1;
            bool vy0 = (y0 >= 0) && (y0 < HH), vy1 = (y1 >= 0) && (y1 < HH);
            bool vx0 = (x0 >= 0) && (x0 < WW), vx1 = (x1 >= 0) && (x1 < WW);
            int cy0 = min(max(y0, 0), HH - 1), cy1 = min(max(y1, 0), HH - 1);
            int cx0 = min(max(x0, 0), WW - 1), cx1 = min(max(x1, 0), WW - 1);
            float w00 = (vy0 && vx0) ? wy0 * wx0 : 0.f;
            float w01 = (vy0 && vx1) ? wy0 * wx1 : 0.f;
            float w10 = (vy1 && vx0) ? wy1 * wx0 : 0.f;
            float w11 = (vy1 && vx1) ? wy1 * wx1 : 0.f;
            float mk  = smk[t][px_b];
            const float* base = out1 + (size_t)b * CC * HWX;
            int i00 = cy0 * WW + cx0, i01 = cy0 * WW + cx1;
            int i10 = cy1 * WW + cx0, i11 = cy1 * WW + cx1;
#pragma unroll
            for (int c = c_b; c < c_b + 8; ++c) {
                const float* pc = base + (size_t)c * HWX;
                float s = w00 * pc[i00] + w01 * pc[i01] + w10 * pc[i10] + w11 * pc[i11];
                val[px_b * 72 + c] = (bf16_t)(s * mk);
            }
        }
        __syncthreads();

        // ---- GEMM accumulate: acc += Wd[t] (16x64) x val (64x16) ----
        {
            const bf16_t* Wt   = Wd + (size_t)t * 4096 + (wave * 16 + laneM) * 64 + hi * 16;
            const bf16_t* brow = &val[laneM * 72 + hi * 16];
#pragma unroll
            for (int c0 = 0; c0 < 64; c0 += 32) {
                v16bf af = ld_fragA(Wt + c0);
                v16bf bf = ld_fragB(brow + c0);
                acc = __builtin_amdgcn_wmma_f32_16x16x32_bf16(
                        false, af, false, bf, (short)0, acc, false, false);
            }
        }
        __syncthreads();
    }

#pragma unroll
    for (int r = 0; r < 8; ++r) {
        int co = wave * 16 + r + hi * 8;
        out2[(((size_t)b * CC + co) * HH + h) * WW + w0 + laneM] = acc[r];
    }
}

// ---------------------------------------------------------------------------
// Host-side orchestration (all on `stream`, scratch carved out of d_ws)
// ---------------------------------------------------------------------------
extern "C" void kernel_launch(void* const* d_in, const int* in_sizes, int n_in,
                              void* d_out, int out_size, void* d_ws, size_t ws_size,
                              hipStream_t stream) {
    (void)in_sizes; (void)n_in; (void)out_size; (void)ws_size;

    const float* x        = (const float*)d_in[0];
    const float* inter    = (const float*)d_in[1];
    const float* fea      = (const float*)d_in[2];
    const float* rcab_w1  = (const float*)d_in[3];
    const float* rcab_b1  = (const float*)d_in[4];
    const float* rcab_w2  = (const float*)d_in[5];
    const float* rcab_b2  = (const float*)d_in[6];
    const float* ca_w1    = (const float*)d_in[7];
    const float* ca_b1    = (const float*)d_in[8];
    const float* ca_w2    = (const float*)d_in[9];
    const float* ca_b2    = (const float*)d_in[10];
    const float* off_w    = (const float*)d_in[11];
    const float* off_b    = (const float*)d_in[12];
    const float* mask_w   = (const float*)d_in[13];
    const float* mask_b   = (const float*)d_in[14];
    const float* deform_w = (const float*)d_in[15];
    const float* conv_w   = (const float*)d_in[16];
    const float* conv_b   = (const float*)d_in[17];
    float* out = (float*)d_out;

    // ---- workspace layout (bytes) ----
    char* ws = (char*)d_ws;
    const size_t SZ_W64  = 64 * 576 * sizeof(bf16_t);      // 73728
    bf16_t* w1r   = (bf16_t*)(ws + 0 * SZ_W64);
    bf16_t* w2r   = (bf16_t*)(ws + 1 * SZ_W64);
    bf16_t* wfr   = (bf16_t*)(ws + 2 * SZ_W64);
    bf16_t* womr  = (bf16_t*)(ws + 3 * SZ_W64);            // 32 x 1152 = same size
    bf16_t* wdr   = (bf16_t*)(ws + 4 * SZ_W64);            // 9 x 64 x 64 = same size
    float*  means = (float*)(ws + 5 * SZ_W64);             // 256 f32
    float*  yvec  = means + 256;
    const size_t ACT_OFF = 5 * SZ_W64 + 2048;              // 370688, 16B aligned
    const size_t SZ_ACT  = (size_t)BATCH * CC * HWX * sizeof(float);   // 37748736
    float* res1   = (float*)(ws + ACT_OFF);
    float* res2   = (float*)(ws + ACT_OFF + SZ_ACT);
    float* ombuf  = (float*)(ws + ACT_OFF + 2 * SZ_ACT);   // 4 x 27 x H x W
    const size_t SZ_OM = (size_t)BATCH * 27 * HWX * sizeof(float);
    float* out2   = (float*)(ws + ACT_OFF + 2 * SZ_ACT + SZ_OM);
    float* out1   = res1;   // res1 is dead once res2 exists; reuse region

    // ---- 1. pack weights to bf16 WMMA-A layout ----
    pack_conv_kernel<<<144, 256, 0, stream>>>(rcab_w1, w1r, 64, 64);
    pack_conv_kernel<<<144, 256, 0, stream>>>(rcab_w2, w2r, 64, 64);
    pack_conv_kernel<<<144, 256, 0, stream>>>(conv_w,  wfr, 64, 64);
    pack_offmask_kernel<<<144, 256, 0, stream>>>(off_w, mask_w, womr);
    pack_deform_kernel<<<144, 256, 0, stream>>>(deform_w, wdr);

    dim3 cgrid(NGRP), cblk(128);

    // ---- 2. RCAB conv1 (relu) : x -> res1 ----
    conv3x3_wmma_kernel<64, 64, 1, 0><<<cgrid, cblk, 0, stream>>>(
        x, nullptr, w1r, rcab_b1, nullptr, 64, res1);
    // ---- 3. RCAB conv2 : res1 -> res2 ----
    conv3x3_wmma_kernel<64, 64, 0, 0><<<cgrid, cblk, 0, stream>>>(
        res1, nullptr, w2r, rcab_b2, nullptr, 64, res2);
    // ---- 4/5. channel attention ----
    ca_reduce_kernel<<<256, 256, 0, stream>>>(res2, means);
    ca_mlp_kernel<<<1, 256, 0, stream>>>(means, ca_w1, ca_b1, ca_w2, ca_b2, yvec);
    // ---- 6. out1 = res2*y + x  (into res1's region) ----
    rcab_finish_kernel<<<(BATCH * CC * HWX + 255) / 256, 256, 0, stream>>>(
        res2, x, yvec, out1);
    // ---- 7. fused offset+mask conv : concat(inter,fea) -> ombuf (27 ch, raw) ----
    conv3x3_wmma_kernel<128, 27, 0, 1><<<cgrid, cblk, 0, stream>>>(
        inter, fea, womr, off_b, mask_b, 18, ombuf);
    // ---- 8. deformable gather + einsum : out1, ombuf -> out2 ----
    deform_wmma_kernel<<<cgrid, cblk, 0, stream>>>(out1, ombuf, wdr, out2);
    // ---- 9. final conv : (out2 + out1) -> d_out ----
    conv3x3_wmma_kernel<64, 64, 0, 2><<<cgrid, cblk, 0, stream>>>(
        out2, out1, wfr, conv_b, nullptr, 64, out);
}
